// DistanceModel1_Trace_22617297780838
// MI455X (gfx1250) — matrix operand order repaired and checked by reading, not compiled, not run on hardware
//
#include <hip/hip_runtime.h>
#include <hip/hip_bf16.h>
#include <cstdint>
#include <cstddef>

typedef _Float16 v8h  __attribute__((ext_vector_type(8)));
typedef _Float16 v16h __attribute__((ext_vector_type(16)));
typedef float    v8f  __attribute__((ext_vector_type(8)));
typedef float    v2f  __attribute__((ext_vector_type(2)));

#define NSAMP 65536
#define QDIM  256
#define PI_F  3.14159265358979f

// ---------------------------------------------------------------------------
// Kernel 1: MLP + ZZ-feature-map embedding.
// One thread = one sample. Writes C,D in basis-major layout: Ct[b][s], Dt[b][s]
// (f16).  psi = C + i D,  C = cos(phi/2)/16,  D = -sin(phi/2)/16.
// ---------------------------------------------------------------------------
__global__ __launch_bounds__(256) void embed_kernel(
    const float* __restrict__ x,
    const float* __restrict__ W1, const float* __restrict__ b1,
    const float* __restrict__ W2, const float* __restrict__ b2,
    const float* __restrict__ W3, const float* __restrict__ b3,
    _Float16* __restrict__ Ct, _Float16* __restrict__ Dt)
{
    const int gid = blockIdx.x * 256 + threadIdx.x;   // sample index

    float xi[8];
    #pragma unroll
    for (int i = 0; i < 8; i++) xi[i] = x[(size_t)gid * 8 + i];

    float h1[10];
    #pragma unroll
    for (int j = 0; j < 10; j++) {
        float s = b1[j];
        #pragma unroll
        for (int i = 0; i < 8; i++) s = fmaf(xi[i], W1[j * 8 + i], s);
        h1[j] = fmaxf(s, 0.0f);
    }
    float h2[10];
    #pragma unroll
    for (int j = 0; j < 10; j++) {
        float s = b2[j];
        #pragma unroll
        for (int i = 0; i < 10; i++) s = fmaf(h1[i], W2[j * 10 + i], s);
        h2[j] = fmaxf(s, 0.0f);
    }
    float h[8];
    #pragma unroll
    for (int j = 0; j < 8; j++) {
        float s = b3[j];
        #pragma unroll
        for (int i = 0; i < 10; i++) s = fmaf(h2[i], W3[j * 10 + i], s);
        h[j] = s;
    }

    float zz[7];
    #pragma unroll
    for (int i = 0; i < 7; i++) zz[i] = (PI_F - h[i]) * (PI_F - h[i + 1]);

    // phase(b) = sum_i h_i*s_i(b) + sum_i zz_i*s_i(b)s_{i+1}(b),
    // s_i = 1-2*bit_{7-i}.  phase = T - 2*acc(b).
    float T = 0.0f;
    #pragma unroll
    for (int i = 0; i < 8; i++) T += h[i];
    #pragma unroll
    for (int i = 0; i < 7; i++) T += zz[i];

    for (int b = 0; b < QDIM; b++) {
        float acc = 0.0f;
        #pragma unroll
        for (int i = 0; i < 8; i++)
            if ((b >> (7 - i)) & 1) acc += h[i];
        #pragma unroll
        for (int i = 0; i < 7; i++)
            if (((b >> (7 - i)) ^ (b >> (6 - i))) & 1) acc += zz[i];
        float ph = 0.5f * T - acc;          // 0.5*(T - 2*acc)
        float sn, cs;
        __sincosf(ph, &sn, &cs);
        Ct[(size_t)b * NSAMP + gid] = (_Float16)(cs * 0.0625f);
        Dt[(size_t)b * NSAMP + gid] = (_Float16)(-sn * 0.0625f);
    }
}

// ---------------------------------------------------------------------------
// Kernel 2: Gram GEMMs via v_wmma_f32_16x16x32_f16.
//   combo 0: Gc += scale * C^T C    combo 1: Gd += scale * D^T D
//   combo 2: Gx += scale * D^T C
// Both A and B fragments load from the basis-major [256][65536] layout:
//   A[m][k] = Mat[m][k]  (rows),  B[k][n] = Mat[n][k] (contiguous K per lane).
// Grid: combo(3) x tile128x64(8) x ksplit(16) = 384 blocks of 256 threads.
// Each wave register-blocks a 2x2 grid of 16x16 subtiles:
//   2 A-frags + 2 B-frags -> 4 WMMAs  (2 b128 loads per WMMA).
// ---------------------------------------------------------------------------
__global__ __launch_bounds__(256) void gram_kernel(
    const _Float16* __restrict__ Ct, const _Float16* __restrict__ Dt,
    float* __restrict__ Gc, float* __restrict__ Gd, float* __restrict__ Gx,
    float scale)
{
    const int bid   = blockIdx.x;
    const int ks    = bid & 15;
    const int tile  = (bid >> 4) & 7;
    const int combo = bid >> 7;            // 0..2

    const _Float16* Af;
    const _Float16* Bf;
    float* Out;
    if (combo == 0)      { Af = Ct; Bf = Ct; Out = Gc; }
    else if (combo == 1) { Af = Dt; Bf = Dt; Out = Gd; }
    else                 { Af = Dt; Bf = Ct; Out = Gx; }

    const int wave = threadIdx.x >> 5;
    const int lane = threadIdx.x & 31;
    const int tr   = tile >> 2;            // 0..1  (128-row tile)
    const int tc   = tile & 3;             // 0..3  (64-col tile)
    const int r2   = wave >> 1;            // 0..3  (pair of row subtiles)
    const int c2   = wave & 1;             // 0..1  (pair of col subtiles)

    const int rowbase0 = tr * 128 + r2 * 32;
    const int rowbase1 = rowbase0 + 16;
    const int colbase0 = tc * 64 + c2 * 32;
    const int colbase1 = colbase0 + 16;

    const int lhi = lane >> 4;             // 0 or 1
    const int ll  = lane & 15;

    const size_t abase0 = (size_t)(rowbase0 + ll) * NSAMP + (size_t)(lhi << 3);
    const size_t abase1 = (size_t)(rowbase1 + ll) * NSAMP + (size_t)(lhi << 3);
    const size_t bbase0 = (size_t)(colbase0 + ll) * NSAMP + (size_t)(lhi << 3);
    const size_t bbase1 = (size_t)(colbase1 + ll) * NSAMP + (size_t)(lhi << 3);

    v8f acc00 = {};
    v8f acc01 = {};
    v8f acc10 = {};
    v8f acc11 = {};

    const int k0 = ks * (NSAMP / 16);
    #pragma unroll 2
    for (int kk = k0; kk < k0 + NSAMP / 16; kk += 32) {
        // software prefetch of the streamed panels (global_prefetch_b8)
        __builtin_prefetch(Af + abase0 + kk + 512, 0, 1);
        __builtin_prefetch(Af + abase1 + kk + 512, 0, 1);
        __builtin_prefetch(Bf + bbase0 + kk + 512, 0, 1);
        __builtin_prefetch(Bf + bbase1 + kk + 512, 0, 1);

        // A fragments (16x32 f16): lane-half 0 -> K 0..7 & 16..23, half 1 -> +8.
        v8h a0l = *(const v8h*)(Af + abase0 + kk);
        v8h a0h = *(const v8h*)(Af + abase0 + kk + 16);
        v16h a0 = __builtin_shufflevector(a0l, a0h, 0,1,2,3,4,5,6,7,8,9,10,11,12,13,14,15);
        v8h a1l = *(const v8h*)(Af + abase1 + kk);
        v8h a1h = *(const v8h*)(Af + abase1 + kk + 16);
        v16h a1 = __builtin_shufflevector(a1l, a1h, 0,1,2,3,4,5,6,7,8,9,10,11,12,13,14,15);
        // B fragments (32x16 f16): lane-half 0 -> K 0..7 & 16..23, half 1 -> +8.
        v8h b0l = *(const v8h*)(Bf + bbase0 + kk);
        v8h b0h = *(const v8h*)(Bf + bbase0 + kk + 16);
        v16h b0 = __builtin_shufflevector(b0l, b0h, 0,1,2,3,4,5,6,7,8,9,10,11,12,13,14,15);
        v8h b1l = *(const v8h*)(Bf + bbase1 + kk);
        v8h b1h = *(const v8h*)(Bf + bbase1 + kk + 16);
        v16h b1 = __builtin_shufflevector(b1l, b1h, 0,1,2,3,4,5,6,7,8,9,10,11,12,13,14,15);

        acc00 = __builtin_amdgcn_wmma_f32_16x16x32_f16(false, a0, false, b0, (short)0, acc00, false, false);
        acc01 = __builtin_amdgcn_wmma_f32_16x16x32_f16(false, a0, false, b1, (short)0, acc01, false, false);
        acc10 = __builtin_amdgcn_wmma_f32_16x16x32_f16(false, a1, false, b0, (short)0, acc10, false, false);
        acc11 = __builtin_amdgcn_wmma_f32_16x16x32_f16(false, a1, false, b1, (short)0, acc11, false, false);
    }

    // C/D layout: VGPR r, lane l -> M = r + 8*(l>>4), N = l&15.
    const int Mofs = lhi << 3;
    const int N    = ll;
    #pragma unroll
    for (int r = 0; r < 8; r++) {
        int Mr0 = rowbase0 + r + Mofs;
        int Mr1 = rowbase1 + r + Mofs;
        atomicAdd(&Out[Mr0 * QDIM + colbase0 + N], acc00[r] * scale);
        atomicAdd(&Out[Mr0 * QDIM + colbase1 + N], acc01[r] * scale);
        atomicAdd(&Out[Mr1 * QDIM + colbase0 + N], acc10[r] * scale);
        atomicAdd(&Out[Mr1 * QDIM + colbase1 + N], acc11[r] * scale);
    }
}

// ---------------------------------------------------------------------------
// Kernel 3: assemble 512x512 real-symmetric embedding of rho_diff:
//   M = [[Re, -Im], [Im, Re]],  Re = Gc+Gd,  Im_ij = Gx_ij - Gx_ji.
// ---------------------------------------------------------------------------
__global__ __launch_bounds__(256) void assemble_kernel(
    const float* __restrict__ Gc, const float* __restrict__ Gd,
    const float* __restrict__ Gx, float* __restrict__ M)
{
    int idx = blockIdx.x * 256 + threadIdx.x;      // 0 .. 512*512-1
    int r = idx >> 9, c = idx & 511;
    float v;
    if (r < 256 && c < 256)
        v = Gc[r * 256 + c] + Gd[r * 256 + c];
    else if (r >= 256 && c >= 256)
        v = Gc[(r - 256) * 256 + (c - 256)] + Gd[(r - 256) * 256 + (c - 256)];
    else if (r >= 256)   // lower-left: +Im
        v = Gx[(r - 256) * 256 + c] - Gx[c * 256 + (r - 256)];
    else                 // upper-right: -Im
        v = Gx[(c - 256) * 256 + r] - Gx[r * 256 + (c - 256)];
    M[idx] = v;
}

// ---------------------------------------------------------------------------
// Kernel 4: Frobenius-norm^2 reduction (atomicAdd into scalar).
// ---------------------------------------------------------------------------
__global__ __launch_bounds__(256) void sumsq_kernel(
    const float* __restrict__ A, float* __restrict__ out)
{
    __shared__ float sm[256];
    float p = 0.0f;
    for (int i = blockIdx.x * 256 + threadIdx.x; i < 512 * 512; i += gridDim.x * 256)
        p = fmaf(A[i], A[i], p);
    sm[threadIdx.x] = p;
    __syncthreads();
    for (int s = 128; s > 0; s >>= 1) {
        if (threadIdx.x < s) sm[threadIdx.x] += sm[threadIdx.x + s];
        __syncthreads();
    }
    if (threadIdx.x == 0) atomicAdd(out, sm[0]);
}

// Kernel 5: X0 = M / ||M||_F
__global__ __launch_bounds__(256) void scale_kernel(
    const float* __restrict__ M, const float* __restrict__ sumsq,
    float* __restrict__ X)
{
    int idx = blockIdx.x * 256 + threadIdx.x;
    X[idx] = M[idx] * __frsqrt_rn(*sumsq);
}

// ---------------------------------------------------------------------------
// Kernel 6: 512x512x512 f32 GEMM on v_wmma_f32_16x16x4_f32.
//   mode 0: Out = X * Yin
//   mode 1: Out = 1.5*Xlin - 0.5*(X * Yin)        (Newton-Schulz step)
// 8 waves/block, one 16x16 tile per wave; 1024 tiles -> 128 blocks.
// (Kept at 1 tile/wave: the 40 sequential NS iterations are latency-bound,
//  so maximum concurrent wave-jobs beats load amortization here.)
// ---------------------------------------------------------------------------
__global__ __launch_bounds__(256) void ns_gemm_kernel(
    const float* __restrict__ X, const float* __restrict__ Yin,
    float* __restrict__ Out, const float* __restrict__ Xlin, int mode)
{
    const int g    = blockIdx.x * 8 + (threadIdx.x >> 5);  // tile id 0..1023
    const int lane = threadIdx.x & 31;
    const int m16  = (g >> 5) << 4;
    const int n16  = (g & 31) << 4;

    const int mrow  = m16 + (lane & 15);
    const int ncol  = n16 + (lane & 15);
    const int khalf = (lane >> 4) << 1;      // 0 or 2

    v8f acc = {};
    #pragma unroll 4
    for (int k = 0; k < 512; k += 4) {
        v2f a = *(const v2f*)(X + (size_t)mrow * 512 + k + khalf);
        v2f b;
        b.x = Yin[(size_t)(k + khalf) * 512 + ncol];
        b.y = Yin[(size_t)(k + khalf + 1) * 512 + ncol];
        acc = __builtin_amdgcn_wmma_f32_16x16x4_f32(false, a, false, b, (short)0, acc, false, false);
    }

    const int Mofs = (lane >> 4) << 3;
    const int N    = n16 + (lane & 15);
    #pragma unroll
    for (int r = 0; r < 8; r++) {
        int Mr = m16 + r + Mofs;
        float v = acc[r];
        if (mode) v = fmaf(1.5f, Xlin[(size_t)Mr * 512 + N], -0.5f * v);
        Out[(size_t)Mr * 512 + N] = v;
    }
}

// ---------------------------------------------------------------------------
// Kernel 7: result = -0.25 * sum_ij M_ij * S_ij  ( = -0.5 * sum|eig(rho_diff)| )
// ---------------------------------------------------------------------------
__global__ __launch_bounds__(256) void trace_kernel(
    const float* __restrict__ M, const float* __restrict__ S,
    float* __restrict__ out)
{
    __shared__ float sm[256];
    float p = 0.0f;
    for (int i = blockIdx.x * 256 + threadIdx.x; i < 512 * 512; i += gridDim.x * 256)
        p = fmaf(M[i], S[i], p);
    sm[threadIdx.x] = p;
    __syncthreads();
    for (int s = 128; s > 0; s >>= 1) {
        if (threadIdx.x < s) sm[threadIdx.x] += sm[threadIdx.x + s];
        __syncthreads();
    }
    if (threadIdx.x == 0) atomicAdd(out, -0.25f * sm[0]);
}

// ---------------------------------------------------------------------------
extern "C" void kernel_launch(void* const* d_in, const int* in_sizes, int n_in,
                              void* d_out, int out_size, void* d_ws, size_t ws_size,
                              hipStream_t stream) {
    (void)in_sizes; (void)n_in; (void)out_size; (void)ws_size;
    const float* x1 = (const float*)d_in[0];
    const float* x0 = (const float*)d_in[1];
    const float* W1 = (const float*)d_in[2];
    const float* b1 = (const float*)d_in[3];
    const float* W2 = (const float*)d_in[4];
    const float* b2 = (const float*)d_in[5];
    const float* W3 = (const float*)d_in[6];
    const float* b3 = (const float*)d_in[7];
    float* out = (float*)d_out;

    char* ws = (char*)d_ws;
    _Float16* Ct = (_Float16*)(ws);                         // 32 MB
    _Float16* Dt = (_Float16*)(ws + (size_t)32 * 1024 * 1024);
    size_t go = (size_t)64 * 1024 * 1024;
    float* Gc = (float*)(ws + go);                          // 256 KB each
    float* Gd = (float*)(ws + go + 262144);
    float* Gx = (float*)(ws + go + 2 * 262144);
    float* Mm = (float*)(ws + go + 3 * 262144);             // 1 MB
    float* Xa = (float*)(ws + go + 3 * 262144 + 1 * 1048576);
    float* Yb = (float*)(ws + go + 3 * 262144 + 2 * 1048576);
    float* Xb = (float*)(ws + go + 3 * 262144 + 3 * 1048576);
    float* sq = (float*)(ws + go + 3 * 262144 + 4 * 1048576);

    hipMemsetAsync(Gc, 0, 3 * 262144, stream);
    hipMemsetAsync(sq, 0, sizeof(float), stream);
    hipMemsetAsync(out, 0, sizeof(float), stream);

    const float invB = 1.0f / (float)NSAMP;

    // dataset 1 (+), dataset 0 (-); C/D buffers reused sequentially.
    embed_kernel<<<NSAMP / 256, 256, 0, stream>>>(x1, W1, b1, W2, b2, W3, b3, Ct, Dt);
    gram_kernel<<<384, 256, 0, stream>>>(Ct, Dt, Gc, Gd, Gx, +invB);
    embed_kernel<<<NSAMP / 256, 256, 0, stream>>>(x0, W1, b1, W2, b2, W3, b3, Ct, Dt);
    gram_kernel<<<384, 256, 0, stream>>>(Ct, Dt, Gc, Gd, Gx, -invB);

    assemble_kernel<<<1024, 256, 0, stream>>>(Gc, Gd, Gx, Mm);
    sumsq_kernel<<<256, 256, 0, stream>>>(Mm, sq);
    scale_kernel<<<1024, 256, 0, stream>>>(Mm, sq, Xa);

    // Newton-Schulz sign iteration: X <- 1.5X - 0.5 X^3  (40 iterations)
    float* Xc = Xa;
    float* Xn = Xb;
    for (int it = 0; it < 40; it++) {
        ns_gemm_kernel<<<128, 256, 0, stream>>>(Xc, Xc, Yb, nullptr, 0); // Y = X*X
        ns_gemm_kernel<<<128, 256, 0, stream>>>(Xc, Yb, Xn, Xc, 1);      // Xn = 1.5X - 0.5 X*Y
        float* t = Xc; Xc = Xn; Xn = t;
    }

    trace_kernel<<<256, 256, 0, stream>>>(Mm, Xc, out);
}
